// GCNFP_54494545052139
// MI455X (gfx1250) — matrix operand order repaired and checked by reading, not compiled, not run on hardware
//
#include <hip/hip_runtime.h>
#include <math.h>

#define NN   131072   // nodes
#define NE   262144   // edges
#define BG   4096     // graphs / batch
#define DIM  256
#define FPD  1024
#define G4   1024     // 4*DIM (LSTM gates)
#define QS   512      // 2*DIM (q_star)
#define INF_ 15       // input feature dim
#define BN_EPS 1e-5f

typedef __attribute__((ext_vector_type(16))) __bf16 v16bf;
typedef __attribute__((ext_vector_type(8)))  float  v8f;

// ---------- helpers ----------
__device__ __forceinline__ unsigned short f2bf(float x) {
  unsigned int u = __float_as_uint(x);
  unsigned int r = u + 0x7FFFu + ((u >> 16) & 1u);   // round-to-nearest-even
  return (unsigned short)(r >> 16);
}
__device__ __forceinline__ float sigmoidf_(float x) { return 1.0f / (1.0f + expf(-x)); }

// ---------- fp32 -> bf16 conversion ----------
__global__ void cvt_bf16_kernel(const float* __restrict__ src,
                                unsigned short* __restrict__ dst, int n) {
  int i = blockIdx.x * 256 + threadIdx.x;
  if (i < n) dst[i] = f2bf(src[i]);
}

// ---------- generic bf16 WMMA GEMM: C[M,N] (+)= A[M,K] @ W[N,K]^T ----------
// One wave computes one 16x16 tile of C, looping K in steps of 32.
// Fragment layouts per CDNA5 ISA 7.12.2 (16-bit A 16x32, 16-bit B 32x16, f32 C/D 16x16).
template <int ACCUM>
__global__ __launch_bounds__(256) void gemm_bf16_wmma(
    const unsigned short* __restrict__ A, const unsigned short* __restrict__ W,
    float* __restrict__ C, int M, int N, int K) {
  const int wave = threadIdx.x >> 5;
  const int lane = threadIdx.x & 31;
  const int tilesN = N >> 4;
  const int totalTiles = (M >> 4) * tilesN;
  const int tile = blockIdx.x * 8 + wave;
  if (tile >= totalTiles) return;               // wave-uniform -> EXEC stays all-1s
  const int tm = tile / tilesN;
  const int tn = tile - tm * tilesN;
  const int lo = lane & 15;
  const int hi = lane >> 4;
  const unsigned short* arow = A + (size_t)(tm * 16 + lo) * K;
  const unsigned short* wrow = W + (size_t)(tn * 16 + lo) * K;
  v8f acc = {};
  for (int k0 = 0; k0 < K; k0 += 32) {
    union { uint4 u[2]; v16bf v; } af, bf;
    // A 16x32 bf16: lanes0-15 row M=lo holds K[0..7]&K[16..23]; lanes16-31 K[8..15]&K[24..31]
    af.u[0] = *(const uint4*)(arow + k0 + 8 * hi);
    af.u[1] = *(const uint4*)(arow + k0 + 16 + 8 * hi);
    // B 32x16 bf16 (B[k][n] = W[n][k]): lanes0-15 N=lo hold K[0..7]&K[16..23]; lanes16-31 K[8..15]&K[24..31]
    bf.u[0] = *(const uint4*)(wrow + k0 + 8 * hi);
    bf.u[1] = *(const uint4*)(wrow + k0 + 16 + 8 * hi);
    acc = __builtin_amdgcn_wmma_f32_16x16x32_bf16(false, af.v, false, bf.v,
                                                  (short)0, acc, false, false);
  }
  // C/D layout: VGPR r, lane -> row = tm*16 + r + 8*hi, col = tn*16 + lo
  float* c0 = C + (size_t)(tm * 16 + 8 * hi) * N + tn * 16 + lo;
#pragma unroll
  for (int r = 0; r < 8; r++) {
    float* p = c0 + (size_t)r * N;
    if (ACCUM) *p += acc[r]; else *p = acc[r];
  }
}

// ---------- lin0: h0 = relu(x @ lin0_w.T + b) -> bf16 ----------
__global__ void lin0_kernel(const float* __restrict__ x, const float* __restrict__ w,
                            const float* __restrict__ b, unsigned short* __restrict__ h0) {
  int node = blockIdx.x;
  int j = threadIdx.x;  // 0..255
  const float* xr = x + (size_t)node * INF_;
  const float* wr = w + (size_t)j * INF_;
  float acc = b[j];
#pragma unroll
  for (int k = 0; k < INF_; k++) acc += xr[k] * wr[k];
  h0[(size_t)node * DIM + j] = f2bf(fmaxf(acc, 0.0f));
}

// ---------- segment pointers from sorted batch (CSR) ----------
__global__ void segptr_kernel(const int* __restrict__ batch, int* __restrict__ segptr) {
  int t = blockIdx.x * 256 + threadIdx.x;
  if (t > BG) return;
  int lo = 0, hi = NN;
  while (lo < hi) { int mid = (lo + hi) >> 1; if (batch[mid] < t) lo = mid + 1; else hi = mid; }
  segptr[t] = lo;
}

// ---------- degree / dinv ----------
__global__ void deg_init_kernel(float* __restrict__ deg) {
  int i = blockIdx.x * 256 + threadIdx.x;
  if (i < NN) deg[i] = 1.0f;  // self loop
}
__global__ void deg_count_kernel(const int* __restrict__ ei, float* __restrict__ deg) {
  int e = blockIdx.x * 256 + threadIdx.x;
  if (e < NE) atomicAdd(&deg[ei[NE + e]], 1.0f);  // col = edge_index[1]
}
__global__ void dinv_kernel(const float* __restrict__ deg, float* __restrict__ dinv) {
  int i = blockIdx.x * 256 + threadIdx.x;
  if (i < NN) dinv[i] = rsqrtf(deg[i]);
}

// ---------- edge scatter: agg[col] += xw[row] * dinv[row]*dinv[col] ----------
__global__ void scatter_kernel(const int* __restrict__ ei, const float* __restrict__ xw,
                               const float* __restrict__ dinv, float* __restrict__ agg) {
  int e = blockIdx.x * 8 + (threadIdx.x >> 5);
  if (e >= NE) return;
  int lane = threadIdx.x & 31;
  int r = ei[e], c = ei[NE + e];
  float nrm = dinv[r] * dinv[c];
  const float* src = xw + (size_t)r * DIM;
  float* dst = agg + (size_t)c * DIM;
#pragma unroll
  for (int j = 0; j < 8; j++)
    atomicAdd(&dst[lane + 32 * j], src[lane + 32 * j] * nrm);
}

// ---------- GCN epilogue: hg = relu(agg + xw*dinv^2 + b) (in place over agg) ----------
__global__ void gcn_final_kernel(float* __restrict__ agg, const float* __restrict__ xw,
                                 const float* __restrict__ dinv, const float* __restrict__ b) {
  size_t idx = (size_t)blockIdx.x * 256 + threadIdx.x;
  int i = (int)(idx >> 8), d = (int)(idx & 255);
  float di = dinv[i];
  float v = agg[idx] + xw[idx] * di * di + b[d];
  agg[idx] = fmaxf(v, 0.0f);
}

// ---------- fingerprint epilogue: BN(eval) + ELU ----------
__global__ void fp_epilogue_kernel(const float* __restrict__ g, const float* __restrict__ bias,
                                   const float* __restrict__ gam, const float* __restrict__ bet,
                                   const float* __restrict__ mean, const float* __restrict__ var,
                                   float* __restrict__ outfp) {
  size_t idx = (size_t)blockIdx.x * 256 + threadIdx.x;
  int d = (int)(idx & 255);
  float t = (g[idx] + bias[d] - mean[d]) * rsqrtf(var[d] + BN_EPS) * gam[d] + bet[d];
  outfp[idx] = t > 0.0f ? t : (expf(t) - 1.0f);
}

// ---------- LSTM cell: gates [BG,1024] -> c,h (f32 + bf16) ----------
__global__ void lstm_cell_kernel(const float* __restrict__ gates, const float* __restrict__ bih,
                                 const float* __restrict__ bhh, float* __restrict__ c,
                                 float* __restrict__ hf, unsigned short* __restrict__ hbf) {
  size_t idx = (size_t)blockIdx.x * 256 + threadIdx.x;
  int g = (int)(idx >> 8), d = (int)(idx & 255);
  const float* gr = gates + (size_t)g * G4;
  float gi = gr[d]       + bih[d]       + bhh[d];
  float gf = gr[256 + d] + bih[256 + d] + bhh[256 + d];
  float gg = gr[512 + d] + bih[512 + d] + bhh[512 + d];
  float go = gr[768 + d] + bih[768 + d] + bhh[768 + d];
  float cn = sigmoidf_(gf) * c[idx] + sigmoidf_(gi) * tanhf(gg);
  float hn = sigmoidf_(go) * tanhf(cn);
  c[idx] = cn; hf[idx] = hn; hbf[idx] = f2bf(hn);
}

// ---------- set2set attention: one wave per graph ----------
__global__ void attention_kernel(const float* __restrict__ hg, const float* __restrict__ hq,
                                 const int* __restrict__ segptr, float* __restrict__ qstar,
                                 unsigned short* __restrict__ qstar_bf) {
  int g = blockIdx.x;
  int lane = threadIdx.x;  // 0..31
  int s0 = segptr[g], s1 = segptr[g + 1];
  float q[8];
#pragma unroll
  for (int j = 0; j < 8; j++) q[j] = hq[(size_t)g * DIM + lane + 32 * j];
  // pass 1: segment max of e_i = dot(x_i, q)
  float mx = -INFINITY;
  for (int i = s0; i < s1; i++) {
    const float* xr = hg + (size_t)i * DIM;
    float p = 0.0f;
#pragma unroll
    for (int j = 0; j < 8; j++) p += xr[lane + 32 * j] * q[j];
#pragma unroll
    for (int m = 16; m; m >>= 1) p += __shfl_xor(p, m, 32);
    mx = fmaxf(mx, p);
  }
  // pass 2: softmax-weighted sum
  float s = 0.0f, r[8];
#pragma unroll
  for (int j = 0; j < 8; j++) r[j] = 0.0f;
  for (int i = s0; i < s1; i++) {
    const float* xr = hg + (size_t)i * DIM;
    float p = 0.0f;
#pragma unroll
    for (int j = 0; j < 8; j++) p += xr[lane + 32 * j] * q[j];
#pragma unroll
    for (int m = 16; m; m >>= 1) p += __shfl_xor(p, m, 32);
    float wgt = expf(p - mx);
    s += wgt;
#pragma unroll
    for (int j = 0; j < 8; j++) r[j] += wgt * xr[lane + 32 * j];
  }
  float inv = (s > 0.0f) ? (1.0f / s) : 0.0f;
  float* qo = qstar + (size_t)g * QS;
  unsigned short* qb = qstar_bf + (size_t)g * QS;
#pragma unroll
  for (int j = 0; j < 8; j++) {
    float rv = r[j] * inv;
    int d = lane + 32 * j;
    qo[d] = q[j];       qo[DIM + d] = rv;
    qb[d] = f2bf(q[j]); qb[DIM + d] = f2bf(rv);
  }
}

// ---------- final: out = [relu(lin1out+b1), out_fp] @ lin2_w.T + lin2_b ----------
__global__ void final_kernel(const float* __restrict__ lin1out, const float* __restrict__ b1,
                             const float* __restrict__ outfp, const float* __restrict__ w2,
                             const float* __restrict__ b2, float* __restrict__ out) {
  int g = blockIdx.x;
  int lane = threadIdx.x;  // 0..31
  float a0 = 0.0f, a1 = 0.0f;
  for (int j = lane; j < DIM; j += 32) {
    float v = fmaxf(lin1out[(size_t)g * DIM + j] + b1[j], 0.0f);
    a0 += v * w2[j];
    a1 += v * w2[QS + j];
  }
  for (int j = lane; j < DIM; j += 32) {
    float v = outfp[(size_t)g * DIM + j];
    a0 += v * w2[DIM + j];
    a1 += v * w2[QS + DIM + j];
  }
#pragma unroll
  for (int m = 16; m; m >>= 1) { a0 += __shfl_xor(a0, m, 32); a1 += __shfl_xor(a1, m, 32); }
  if (lane == 0) { out[g * 2 + 0] = a0 + b2[0]; out[g * 2 + 1] = a1 + b2[1]; }
}

// =======================================================================
extern "C" void kernel_launch(void* const* d_in, const int* in_sizes, int n_in,
                              void* d_out, int out_size, void* d_ws, size_t ws_size,
                              hipStream_t stream) {
  (void)in_sizes; (void)n_in; (void)out_size; (void)ws_size;
  const float* x      = (const float*)d_in[0];
  const int*   ei     = (const int*)d_in[1];
  const int*   batch  = (const int*)d_in[2];
  const float* fp     = (const float*)d_in[3];
  const float* fc1_w  = (const float*)d_in[4];
  const float* fc1_b  = (const float*)d_in[5];
  const float* bn_g   = (const float*)d_in[6];
  const float* bn_b   = (const float*)d_in[7];
  const float* bn_m   = (const float*)d_in[8];
  const float* bn_v   = (const float*)d_in[9];
  const float* lin0_w = (const float*)d_in[10];
  const float* lin0_b = (const float*)d_in[11];
  const float* gcn_w  = (const float*)d_in[12];
  const float* gcn_b  = (const float*)d_in[13];
  const float* wih    = (const float*)d_in[14];
  const float* whh    = (const float*)d_in[15];
  const float* bih    = (const float*)d_in[16];
  const float* bhh    = (const float*)d_in[17];
  const float* lin1_w = (const float*)d_in[18];
  const float* lin1_b = (const float*)d_in[19];
  const float* lin2_w = (const float*)d_in[20];
  const float* lin2_b = (const float*)d_in[21];
  float* out = (float*)d_out;

  // workspace carve-up
  char* wp = (char*)d_ws;
  auto nxt = [&](size_t bytes) -> void* {
    void* p = (void*)wp; wp += (bytes + 255) & ~(size_t)255; return p;
  };
  unsigned short* fp_bf    = (unsigned short*)nxt((size_t)BG * FPD * 2);
  unsigned short* fc1w_bf  = (unsigned short*)nxt((size_t)DIM * FPD * 2);
  unsigned short* gcnw_bf  = (unsigned short*)nxt((size_t)DIM * DIM * 2);
  unsigned short* wih_bf   = (unsigned short*)nxt((size_t)G4 * QS * 2);
  unsigned short* whh_bf   = (unsigned short*)nxt((size_t)G4 * DIM * 2);
  unsigned short* lin1w_bf = (unsigned short*)nxt((size_t)DIM * QS * 2);
  unsigned short* h0_bf    = (unsigned short*)nxt((size_t)NN * DIM * 2);
  float* xw       = (float*)nxt((size_t)NN * DIM * 4);
  float* agg      = (float*)nxt((size_t)NN * DIM * 4);   // becomes hg in place
  float* deg      = (float*)nxt((size_t)NN * 4);
  float* dinv     = (float*)nxt((size_t)NN * 4);
  int*   segptr   = (int*)nxt((size_t)(BG + 1) * 4);
  float* fpg      = (float*)nxt((size_t)BG * DIM * 4);
  float* outfp    = (float*)nxt((size_t)BG * DIM * 4);
  float* gates    = (float*)nxt((size_t)BG * G4 * 4);
  float* cbuf     = (float*)nxt((size_t)BG * DIM * 4);
  float* hf       = (float*)nxt((size_t)BG * DIM * 4);
  unsigned short* hbf      = (unsigned short*)nxt((size_t)BG * DIM * 2);
  float* qstar    = (float*)nxt((size_t)BG * QS * 4);
  unsigned short* qstar_bf = (unsigned short*)nxt((size_t)BG * QS * 2);
  float* lin1out  = (float*)nxt((size_t)BG * DIM * 4);

  // --- weight / input conversions to bf16 ---
  auto cvt = [&](const float* s, unsigned short* d, int n) {
    cvt_bf16_kernel<<<(n + 255) / 256, 256, 0, stream>>>(s, d, n);
  };
  cvt(fp,     fp_bf,    BG * FPD);
  cvt(fc1_w,  fc1w_bf,  DIM * FPD);
  cvt(gcn_w,  gcnw_bf,  DIM * DIM);
  cvt(wih,    wih_bf,   G4 * QS);
  cvt(whh,    whh_bf,   G4 * DIM);
  cvt(lin1_w, lin1w_bf, DIM * QS);

  // --- graph branch ---
  lin0_kernel<<<NN, 256, 0, stream>>>(x, lin0_w, lin0_b, h0_bf);
  gemm_bf16_wmma<0><<<(NN / 16) * (DIM / 16) / 8, 256, 0, stream>>>(
      h0_bf, gcnw_bf, xw, NN, DIM, DIM);

  segptr_kernel<<<(BG + 1 + 255) / 256, 256, 0, stream>>>(batch, segptr);
  deg_init_kernel<<<NN / 256, 256, 0, stream>>>(deg);
  deg_count_kernel<<<NE / 256, 256, 0, stream>>>(ei, deg);
  dinv_kernel<<<NN / 256, 256, 0, stream>>>(deg, dinv);

  hipMemsetAsync(agg, 0, (size_t)NN * DIM * 4, stream);
  scatter_kernel<<<NE / 8, 256, 0, stream>>>(ei, xw, dinv, agg);
  gcn_final_kernel<<<(size_t)NN * DIM / 256, 256, 0, stream>>>(agg, xw, dinv, gcn_b);
  float* hg = agg;  // relu'd GCN output, in place

  // --- fingerprint branch ---
  gemm_bf16_wmma<0><<<(BG / 16) * (DIM / 16) / 8, 256, 0, stream>>>(
      fp_bf, fc1w_bf, fpg, BG, DIM, FPD);
  fp_epilogue_kernel<<<(size_t)BG * DIM / 256, 256, 0, stream>>>(
      fpg, fc1_b, bn_g, bn_b, bn_m, bn_v, outfp);

  // --- set2set (3 steps) ---
  hipMemsetAsync(cbuf,     0, (size_t)BG * DIM * 4, stream);
  hipMemsetAsync(hbf,      0, (size_t)BG * DIM * 2, stream);
  hipMemsetAsync(qstar_bf, 0, (size_t)BG * QS * 2, stream);
  for (int step = 0; step < 3; step++) {
    gemm_bf16_wmma<0><<<(BG / 16) * (G4 / 16) / 8, 256, 0, stream>>>(
        qstar_bf, wih_bf, gates, BG, G4, QS);
    gemm_bf16_wmma<1><<<(BG / 16) * (G4 / 16) / 8, 256, 0, stream>>>(
        hbf, whh_bf, gates, BG, G4, DIM);
    lstm_cell_kernel<<<(size_t)BG * DIM / 256, 256, 0, stream>>>(
        gates, bih, bhh, cbuf, hf, hbf);
    attention_kernel<<<BG, 32, 0, stream>>>(hg, hf, segptr, qstar, qstar_bf);
  }

  // --- heads ---
  gemm_bf16_wmma<0><<<(BG / 16) * (DIM / 16) / 8, 256, 0, stream>>>(
      qstar_bf, lin1w_bf, lin1out, BG, DIM, QS);
  final_kernel<<<BG, 32, 0, stream>>>(lin1out, lin1_b, outfp, lin2_w, lin2_b, out);
}
